// GraphAttentionTravelNet_20779051778570
// MI455X (gfx1250) — compile-verified, matrix operand order checked by reading, not run on hardware
//
#include <hip/hip_runtime.h>
#include <hip/hip_bf16.h>
#include <math.h>

typedef __attribute__((ext_vector_type(16))) __bf16        v16bf;
typedef __attribute__((ext_vector_type(8)))  float         v8f;
typedef __attribute__((ext_vector_type(8)))  unsigned int  v8u;

// ---------------------------------------------------------------------------
// helpers
// ---------------------------------------------------------------------------
__device__ __forceinline__ unsigned f2bf(float f) {
    unsigned u = __float_as_uint(f);
    u += 0x7FFFu + ((u >> 16) & 1u);      // round-to-nearest-even
    return u >> 16;
}
__device__ __forceinline__ unsigned pack_bf2(float lo, float hi) {
    return f2bf(lo) | (f2bf(hi) << 16);
}

// CDNA5 16-bit A/B fragment: lane L -> m(or n) = L%16, half = L/16.
// VGPR v<4 : K = half*8 + 2v ; v>=4 : K = 16 + half*8 + 2(v-4). Rows in LDS
// are K-contiguous so each bf16 pair is one aligned 32-bit LDS read.
__device__ __forceinline__ v16bf load_frag(const unsigned short* rowp, int half) {
    v8u u;
    const unsigned* p0 = (const unsigned*)(rowp + half * 8);
    const unsigned* p1 = (const unsigned*)(rowp + 16 + half * 8);
    u[0] = p0[0]; u[1] = p0[1]; u[2] = p0[2]; u[3] = p0[3];
    u[4] = p1[0]; u[5] = p1[1]; u[6] = p1[2]; u[7] = p1[3];
    return __builtin_bit_cast(v16bf, u);
}

__device__ __forceinline__ void atomicMaxF(float* addr, float val) {
    if (val >= 0.0f) atomicMax((int*)addr, __float_as_int(val));
    else             atomicMin((unsigned int*)addr, __float_as_uint(val));
}

// ---------------------------------------------------------------------------
// Generic fp32-in / fp32-out GEMM, bf16 WMMA compute.  C = A[M,K] @ B[K,N] (+bias)
// Block: 64(M) x 128(N) tile, 128 threads (4 waves); wave tile 32x64
//  = 2 A-frags x 4 B-frags -> 8 x v_wmma_f32_16x16x32_bf16 per K-step.
// K multiple of 32, N multiple of 128; M guarded.
// ---------------------------------------------------------------------------
__global__ __launch_bounds__(128) void gemm_wmma_kernel(
        const float* __restrict__ A, const float* __restrict__ B,
        const float* __restrict__ bias, float* __restrict__ C,
        int M, int K, int N) {
    __shared__ unsigned short As[64][34];    // [m][k] bf16, padded (17 dwords/row)
    __shared__ unsigned short Bs[128][34];   // [n][k] bf16 (B staged transposed)

    const int mBase = blockIdx.x * 64;
    const int nBase = blockIdx.y * 128;
    const int tid   = threadIdx.x;
    const int wave  = tid >> 5, lane = tid & 31;
    const int wm    = wave >> 1, wn = wave & 1;
    const int lm    = lane & 15, lhalf = lane >> 4;

    v8f acc[2][4] = {};

    for (int k0 = 0; k0 < K; k0 += 32) {
        // prefetch next K-slice (speculative; dropped if OOB)
        if (k0 + 32 < K) {
            int pr = mBase + (tid >> 1);
            if (pr < M) __builtin_prefetch(A + (size_t)pr * K + k0 + 32, 0, 1);
            __builtin_prefetch(B + (size_t)(k0 + 32 + (tid >> 2)) * N + nBase + (tid & 3) * 32, 0, 1);
        }
        // stage A tile 64x32 (fp32 -> packed bf16 pairs)
        #pragma unroll
        for (int i = 0; i < 4; i++) {
            int q   = tid * 4 + i;            // 512 float4 = 64 rows x 8
            int row = q >> 3, c4 = (q & 7) * 4;
            float4 v = make_float4(0.f, 0.f, 0.f, 0.f);
            int gr = mBase + row;
            if (gr < M) v = *(const float4*)(A + (size_t)gr * K + k0 + c4);
            *(unsigned*)&As[row][c4 + 0] = pack_bf2(v.x, v.y);
            *(unsigned*)&As[row][c4 + 2] = pack_bf2(v.z, v.w);
        }
        // stage B tile 32x128 transposed to [n][k]; each thread takes a k-pair
        // so (k,k+1) packs into one 32-bit LDS store per destination row.
        #pragma unroll
        for (int i = 0; i < 4; i++) {
            int q   = tid * 4 + i;            // 512 tasks = 16 k-pairs x 32 col4
            int kp  = q >> 5, col = (q & 31) * 4;
            const float* b0 = B + (size_t)(k0 + 2 * kp) * N + nBase + col;
            float4 v0 = *(const float4*)(b0);
            float4 v1 = *(const float4*)(b0 + N);
            *(unsigned*)&Bs[col + 0][2 * kp] = pack_bf2(v0.x, v1.x);
            *(unsigned*)&Bs[col + 1][2 * kp] = pack_bf2(v0.y, v1.y);
            *(unsigned*)&Bs[col + 2][2 * kp] = pack_bf2(v0.z, v1.z);
            *(unsigned*)&Bs[col + 3][2 * kp] = pack_bf2(v0.w, v1.w);
        }
        __syncthreads();

        v16bf af[2], bfr[4];
        #pragma unroll
        for (int t = 0; t < 2; t++) af[t]  = load_frag(&As[wm * 32 + t * 16 + lm][0], lhalf);
        #pragma unroll
        for (int t = 0; t < 4; t++) bfr[t] = load_frag(&Bs[wn * 64 + t * 16 + lm][0], lhalf);

        #pragma unroll
        for (int tm = 0; tm < 2; tm++)
            #pragma unroll
            for (int tn = 0; tn < 4; tn++)
                acc[tm][tn] = __builtin_amdgcn_wmma_f32_16x16x32_bf16(
                    false, af[tm], false, bfr[tn], (short)0, acc[tm][tn], false, false);
        __syncthreads();
    }

    // C layout: vgpr r -> M = r + half*8 ; lane%16 -> N
    #pragma unroll
    for (int tm = 0; tm < 2; tm++)
        #pragma unroll
        for (int tn = 0; tn < 4; tn++) {
            int col  = nBase + wn * 64 + tn * 16 + lm;
            float bv = bias ? bias[col] : 0.0f;
            #pragma unroll
            for (int r = 0; r < 8; r++) {
                int row = mBase + wm * 32 + tm * 16 + r + lhalf * 8;
                if (row < M) C[(size_t)row * N + col] = acc[tm][tn][r] + bv;
            }
        }
}

// ---------------------------------------------------------------------------
// a_dst fold: v[k][h] = sum_c W[k][h*32+c] * adst[h][c]   (256x8 output)
// ---------------------------------------------------------------------------
__global__ __launch_bounds__(256) void fold_adst_kernel(
        const float* __restrict__ W, const float* __restrict__ adst,
        float* __restrict__ v) {
    int idx = blockIdx.x * 256 + threadIdx.x;
    if (idx >= 2048) return;
    int k = idx >> 3, h = idx & 7;
    float acc = 0.f;
    #pragma unroll
    for (int c = 0; c < 32; c++) acc += W[k * 256 + h * 32 + c] * adst[h * 32 + c];
    v[idx] = acc;
}

// ald[n][h] = x_d[n,:] @ v[:,h]
__global__ __launch_bounds__(256) void ald_gemv_kernel(
        const float* __restrict__ xd, const float* __restrict__ v,
        float* __restrict__ ald, int Nd) {
    __shared__ float vs[2048];
    for (int i = threadIdx.x; i < 2048; i += 256) vs[i] = v[i];
    __syncthreads();
    int idx = blockIdx.x * 256 + threadIdx.x;
    if (idx >= Nd * 8) return;
    int n = idx >> 3, h = idx & 7;
    const float* xr = xd + (size_t)n * 256;
    float acc = 0.f;
    for (int k = 0; k < 256; k++) acc += xr[k] * vs[k * 8 + h];
    ald[idx] = acc;
}

// als[n][h] = sum_c hs[n][h*32+c] * asrc[h][c]
__global__ __launch_bounds__(256) void als_reduce_kernel(
        const float* __restrict__ hs, const float* __restrict__ asrc,
        float* __restrict__ als, int Ns) {
    __shared__ float as_s[256];
    if (threadIdx.x < 256) as_s[threadIdx.x] = asrc[threadIdx.x];
    __syncthreads();
    int idx = blockIdx.x * 256 + threadIdx.x;
    if (idx >= Ns * 8) return;
    int n = idx >> 3, h = idx & 7;
    const float* hr = hs + (size_t)n * 256 + h * 32;
    float acc = 0.f;
    #pragma unroll
    for (int c = 0; c < 32; c++) acc += hr[c] * as_s[h * 32 + c];
    als[idx] = acc;
}

__global__ __launch_bounds__(256) void fill_kernel(float* __restrict__ p, float val, int n) {
    int i = blockIdx.x * 256 + threadIdx.x;
    if (i < n) p[i] = val;
}

// phase 1: a = leaky_relu(als[src]+ald[dst]); store; atomic segment-max by dst
__global__ __launch_bounds__(256) void edge_logits_kernel(
        const int* __restrict__ ei, int E,
        const float* __restrict__ als, const float* __restrict__ ald,
        float* __restrict__ ea, float* __restrict__ amax) {
    int e = blockIdx.x * 256 + threadIdx.x;
    if (e >= E) return;
    int s = ei[e], d = ei[E + e];
    const float4* ps = (const float4*)(als + (size_t)s * 8);
    const float4* pd = (const float4*)(ald + (size_t)d * 8);
    float4 s0 = ps[0], s1 = ps[1], d0 = pd[0], d1 = pd[1];
    float a[8] = { s0.x + d0.x, s0.y + d0.y, s0.z + d0.z, s0.w + d0.w,
                   s1.x + d1.x, s1.y + d1.y, s1.z + d1.z, s1.w + d1.w };
    float4 o0, o1;
    #pragma unroll
    for (int h = 0; h < 8; h++) {
        float v = a[h] > 0.f ? a[h] : 0.2f * a[h];
        ((float*)&o0)[h & 3] = v;       // first 4 -> o0 (h<4), else o1
        if (h >= 4) ((float*)&o1)[h - 4] = v;
        else        ((float*)&o0)[h]     = v;
        atomicMaxF(&amax[(size_t)d * 8 + h], v);
    }
    float4* po = (float4*)(ea + (size_t)e * 8);
    po[0] = o0; po[1] = o1;
}

// phase 2: ex = exp(a - amax[dst]); store; atomic segment-sum denominator
__global__ __launch_bounds__(256) void edge_exp_kernel(
        const int* __restrict__ ei, int E,
        float* __restrict__ ea, const float* __restrict__ amax,
        float* __restrict__ den) {
    int e = blockIdx.x * 256 + threadIdx.x;
    if (e >= E) return;
    int d = ei[E + e];
    float4* pe = (float4*)(ea + (size_t)e * 8);
    const float4* pm = (const float4*)(amax + (size_t)d * 8);
    float4 a0 = pe[0], a1 = pe[1], m0 = pm[0], m1 = pm[1];
    float ex[8] = { __expf(a0.x - m0.x), __expf(a0.y - m0.y),
                    __expf(a0.z - m0.z), __expf(a0.w - m0.w),
                    __expf(a1.x - m1.x), __expf(a1.y - m1.y),
                    __expf(a1.z - m1.z), __expf(a1.w - m1.w) };
    pe[0] = make_float4(ex[0], ex[1], ex[2], ex[3]);
    pe[1] = make_float4(ex[4], ex[5], ex[6], ex[7]);
    #pragma unroll
    for (int h = 0; h < 8; h++) atomicAdd(&den[(size_t)d * 8 + h], ex[h]);
}

// phase 3: one wave per edge; lane covers 8 contiguous features (one head each)
__global__ __launch_bounds__(256) void edge_scatter_kernel(
        const int* __restrict__ ei, int E,
        const float* __restrict__ ea, const float* __restrict__ den,
        const float* __restrict__ hs, float* __restrict__ outacc) {
    int wave = threadIdx.x >> 5, lane = threadIdx.x & 31;
    int e = blockIdx.x * 8 + wave;
    if (e >= E) return;
    int s = ei[e], d = ei[E + e];
    int h = lane >> 2;                       // features lane*8..lane*8+7 share head
    float alpha = ea[(size_t)e * 8 + h] / den[(size_t)d * 8 + h];
    const float4* hp = (const float4*)(hs + (size_t)s * 256 + lane * 8);
    float4 v0 = hp[0], v1 = hp[1];
    float* op = outacc + (size_t)d * 256 + lane * 8;
    atomicAdd(op + 0, alpha * v0.x); atomicAdd(op + 1, alpha * v0.y);
    atomicAdd(op + 2, alpha * v0.z); atomicAdd(op + 3, alpha * v0.w);
    atomicAdd(op + 4, alpha * v1.x); atomicAdd(op + 5, alpha * v1.y);
    atomicAdd(op + 6, alpha * v1.z); atomicAdd(op + 7, alpha * v1.w);
}

// per-type summed bias: bsum[t][col] = sum_{e: dst(e)==t} b_gat[l][e][col]
__global__ __launch_bounds__(256) void bias_sum_kernel(
        const float* __restrict__ bgat_l, float* __restrict__ bsum) {
    const int etd[27] = {1,1,1,3,5,4,2,3,5,4,3,5,4,2,1,1,1,1,0,1,3,0,1,3,5,4,2};
    int idx = blockIdx.x * 256 + threadIdx.x;
    if (idx >= 1536) return;
    int t = idx >> 8, col = idx & 255;
    float acc = 0.f;
    #pragma unroll
    for (int e = 0; e < 27; e++)
        if (etd[e] == t) acc += bgat_l[e * 256 + col];
    bsum[idx] = acc;
}

// x = relu(x + out + bsum[col])
__global__ __launch_bounds__(256) void residual_relu_kernel(
        float* __restrict__ x, const float* __restrict__ o,
        const float* __restrict__ bsum256, int n) {
    int i = blockIdx.x * 256 + threadIdx.x;
    if (i >= n) return;
    float v = x[i] + o[i] + bsum256[i & 255];
    x[i] = v > 0.f ? v : 0.f;
}

// ---------------------------------------------------------------------------
// host
// ---------------------------------------------------------------------------
extern "C" void kernel_launch(void* const* d_in, const int* in_sizes, int n_in,
                              void* d_out, int out_size, void* d_ws, size_t ws_size,
                              hipStream_t stream) {
    (void)n_in; (void)out_size; (void)ws_size;
    static const int NN[6]  = {100000, 20000, 80000, 60000, 40000, 60000};
    static const int ES[27] = {0,0,0,0,0,0,0,0,0,0,1,1,1,1,3,5,4,2,0,1,3,0,1,3,5,4,2};
    static const int ED[27] = {1,1,1,3,5,4,2,3,5,4,3,5,4,2,1,1,1,1,0,1,3,0,1,3,5,4,2};

    const float* xin[6];
    for (int i = 0; i < 6; i++) xin[i] = (const float*)d_in[i];
    const int* ei[27]; int Ecnt[27];
    for (int e = 0; e < 27; e++) { ei[e] = (const int*)d_in[6 + e]; Ecnt[e] = in_sizes[6 + e] / 2; }
    const float* W_in  = (const float*)d_in[33];
    const float* b_in  = (const float*)d_in[34];
    const float* W_gat = (const float*)d_in[35];
    const float* a_src = (const float*)d_in[36];
    const float* a_dst = (const float*)d_in[37];
    const float* b_gat = (const float*)d_in[38];
    const float* W_out = (const float*)d_in[39];
    const float* b_out = (const float*)d_in[40];
    float* outf = (float*)d_out;

    size_t xoff[6], ooff[6], acc = 0;
    for (int i = 0; i < 6; i++) { xoff[i] = acc * 256; ooff[i] = acc * 128; acc += NN[i]; }
    const size_t TOT = acc;                     // 360000 nodes

    float* p = (float*)d_ws;
    float* xbuf  = p; p += TOT * 256;           // current activations
    float* outb  = p; p += TOT * 256;           // layer accumulator
    float* hsb   = p; p += (size_t)100000 * 256;
    float* alsb  = p; p += (size_t)100000 * 8;
    float* aldb  = p; p += (size_t)100000 * 8;
    float* amaxb = p; p += (size_t)100000 * 8;
    float* denb  = p; p += (size_t)100000 * 8;
    float* eexb  = p; p += (size_t)400000 * 8;
    float* vfold = p; p += 2048;
    float* bsum  = p; p += 2048;

    auto gemm = [&](const float* A, const float* B, const float* bias, float* C,
                    int M, int K, int N) {
        dim3 g((M + 63) / 64, N / 128);
        gemm_wmma_kernel<<<g, 128, 0, stream>>>(A, B, bias, C, M, K, N);
    };

    // input projections: x[i] = xs[i] @ W_in[i] + b_in[i]
    for (int i = 0; i < 6; i++)
        gemm(xin[i], W_in + (size_t)i * 128 * 256, b_in + i * 256,
             xbuf + xoff[i], NN[i], 128, 256);

    for (int l = 0; l < 3; l++) {
        hipMemsetAsync(outb, 0, TOT * 256 * sizeof(float), stream);
        for (int e = 0; e < 27; e++) {
            const int s = ES[e], d = ED[e];
            const int Ns = NN[s], Nd = NN[d], E = Ecnt[e];
            const float* W = W_gat + (size_t)(l * 27 + e) * 256 * 256;

            gemm(xbuf + xoff[s], W, nullptr, hsb, Ns, 256, 256);
            als_reduce_kernel<<<(Ns * 8 + 255) / 256, 256, 0, stream>>>(
                hsb, a_src + (size_t)(l * 27 + e) * 256, alsb, Ns);
            fold_adst_kernel<<<8, 256, 0, stream>>>(
                W, a_dst + (size_t)(l * 27 + e) * 256, vfold);
            ald_gemv_kernel<<<(Nd * 8 + 255) / 256, 256, 0, stream>>>(
                xbuf + xoff[d], vfold, aldb, Nd);

            fill_kernel<<<(Nd * 8 + 255) / 256, 256, 0, stream>>>(
                amaxb, -__builtin_inff(), Nd * 8);
            hipMemsetAsync(denb, 0, (size_t)Nd * 8 * sizeof(float), stream);

            edge_logits_kernel<<<(E + 255) / 256, 256, 0, stream>>>(
                ei[e], E, alsb, aldb, eexb, amaxb);
            edge_exp_kernel<<<(E + 255) / 256, 256, 0, stream>>>(
                ei[e], E, eexb, amaxb, denb);
            edge_scatter_kernel<<<(E + 7) / 8, 256, 0, stream>>>(
                ei[e], E, eexb, denb, hsb, outb + xoff[d]);
        }
        bias_sum_kernel<<<6, 256, 0, stream>>>(b_gat + (size_t)l * 27 * 256, bsum);
        for (int i = 0; i < 6; i++) {
            int n = NN[i] * 256;
            residual_relu_kernel<<<(n + 255) / 256, 256, 0, stream>>>(
                xbuf + xoff[i], outb + xoff[i], bsum + i * 256, n);
        }
    }

    // output projections
    for (int i = 0; i < 6; i++)
        gemm(xbuf + xoff[i], W_out + (size_t)i * 256 * 128, b_out + i * 128,
             outf + ooff[i], NN[i], 256, 128);
}
